// CTRGCN_77077483094614
// MI455X (gfx1250) — matrix operand order repaired and checked by reading, not compiled, not run on hardware
//
#include <hip/hip_runtime.h>
#include <hip/hip_bf16.h>
#include <cstdint>
#include <cstddef>

// ---------------------------------------------------------------------------
// CTR-GCN forward for MI455X (gfx1250, wave32, WMMA bf16).
// GEMMs on v_wmma_f32_16x16x32_bf16 (fp32 accum). Per block: 128 O-rows x
// 64-col strip; each wave holds 4 accumulators and reuses one A fragment
// across 4 chained WMMAs per K-step. Double-buffered LDS tiles in fragment-
// native layout (2x ds_load_b128 per operand), one barrier per K-step,
// select-based (cndmask) zero-padding with no clamped addressing on the
// activation path (all GEMM activations live in interior workspace slabs).
// All column counts (Tout*17) are exact multiples of 64 for this network.
// ---------------------------------------------------------------------------

typedef __attribute__((ext_vector_type(8)))  __bf16 v8bf;
typedef __attribute__((ext_vector_type(16))) __bf16 v16bf;
typedef __attribute__((ext_vector_type(8)))  float  v8f;

#define VJ 17
#define EPSB 1e-5f

__device__ __forceinline__ __bf16 tobf(float f) { return (__bf16)f; }

// ---------------------------------------------------------------- adjacency
__global__ void adj_k(float* __restrict__ Am) {
  if (threadIdx.x != 0 || blockIdx.x != 0) return;
  float A[VJ][VJ];
  for (int i = 0; i < VJ; ++i)
    for (int j = 0; j < VJ; ++j) A[i][j] = 0.f;
  const int ei[16] = {10, 9, 8, 7, 8, 11, 12, 8, 14, 15, 0, 4, 5, 0, 1, 2};
  const int ej[16] = {9, 8, 7, 0, 11, 12, 13, 14, 15, 16, 4, 5, 6, 1, 2, 3};
  for (int e = 0; e < 16; ++e) { A[ei[e]][ej[e]] = 1.f; A[ej[e]][ei[e]] = 1.f; }
  for (int i = 0; i < VJ; ++i) A[i][i] += 1.f;
  float d[VJ];
  for (int i = 0; i < VJ; ++i) {
    float s = 0.f;
    for (int j = 0; j < VJ; ++j) s += A[i][j];
    d[i] = rsqrtf(s);
  }
  for (int i = 0; i < VJ; ++i)
    for (int j = 0; j < VJ; ++j) Am[i * VJ + j] = d[i] * A[i][j] * d[j];
}

// ---------------------------------------------------------------- data BN
__global__ void data_bn_k(const float* __restrict__ X, const float* __restrict__ g,
                          const float* __restrict__ b, const float* __restrict__ m,
                          const float* __restrict__ v, float* __restrict__ Y,
                          int N, int C, int T) {
  size_t idx = (size_t)blockIdx.x * blockDim.x + threadIdx.x;
  size_t total = (size_t)N * C * T * VJ;
  if (idx >= total) return;
  int vv = (int)(idx % VJ);
  int c = (int)((idx / ((size_t)VJ * T)) % C);
  int ch = c * VJ + vv;
  Y[idx] = (X[idx] - m[ch]) * (g[ch] * rsqrtf(v[ch] + EPSB)) + b[ch];
}

// ---------------------------------------------------------------- mean over T
__global__ void mean_t_k(const float* __restrict__ X, float* __restrict__ XM,
                         int N, int C, int T) {
  int idx = blockIdx.x * blockDim.x + threadIdx.x;
  int total = N * C * VJ;
  if (idx >= total) return;
  int v = idx % VJ;
  int c = (idx / VJ) % C;
  int n = idx / (VJ * C);
  float s = 0.f;
  for (int t = 0; t < T; ++t) s += X[((size_t)(n * C + c) * T + t) * VJ + v];
  XM[idx] = s / (float)T;
}

// ---------------------------------------------------------------- attention
// alpha[n,c,v] = softmax_v( -(w2 @ xm + b2)[n,c,v] )   (row-shift invariance)
__global__ void alpha_k(const float* __restrict__ XM, const float* __restrict__ W2,
                        const float* __restrict__ B2, float* __restrict__ AL,
                        int N, int Cin, int inter) {
  int idx = blockIdx.x * blockDim.x + threadIdx.x;
  if (idx >= N * inter) return;
  int c = idx % inter;
  int n = idx / inter;
  float z[VJ];
  for (int v = 0; v < VJ; ++v) {
    float s = B2[c];
    for (int ci = 0; ci < Cin; ++ci) s += W2[c * Cin + ci] * XM[(n * Cin + ci) * VJ + v];
    z[v] = -s;
  }
  float mx = z[0];
  for (int v = 1; v < VJ; ++v) mx = fmaxf(mx, z[v]);
  float sum = 0.f;
  for (int v = 0; v < VJ; ++v) { z[v] = __expf(z[v] - mx); sum += z[v]; }
  float inv = 1.f / sum;
  for (int v = 0; v < VJ; ++v) AL[(size_t)idx * VJ + v] = z[v] * inv;
}

// ---------------------------------------------------------------- 1x1 conv GEMM (WMMA bf16)
// Y[n,o,to,v] = BN?( sum_c W[o,c]*X[n,c,to*stride,v] + bias[o] )
__global__ __launch_bounds__(256)
void conv1x1_wmma(const float* __restrict__ X, const float* __restrict__ W,
                  const float* __restrict__ bias,
                  const float* __restrict__ bg, const float* __restrict__ bb,
                  const float* __restrict__ bm, const float* __restrict__ bv,
                  float* __restrict__ Y,
                  int N, int Cin, int Tin, int Tout, int O, int stride, int Mtiles) {
  __shared__ __attribute__((aligned(64))) __bf16 At[2][128][40];
  __shared__ __attribute__((aligned(64))) __bf16 Bt[2][64][40];
  const int tid = threadIdx.x, lane = tid & 31, wave = tid >> 5;
  const int n = blockIdx.x / Mtiles;
  const int m0 = (blockIdx.x % Mtiles) * 64;
  const int o0 = blockIdx.y * 128;
  const int M = Tout * VJ;  // exact multiple of 64 for this network
  const int arow = (wave << 4) + (lane & 15);
  const int kh = (lane & 16) ? 8 : 0;
  const int bkb = (lane & 16) ? 8 : 0;
  const int bcol = lane & 15;
  const int wr = tid >> 1;         // weight row 0..127
  const int wkb = (tid & 1) * 16;  // weight K sub-block
  const int sk = tid >> 3;         // act K row 0..31
  const int scol = (tid & 7) * 8;  // act col octet
  const int steps = (Cin + 31) >> 5;

  int rg = o0 + wr;
  if (rg > O - 1) rg = O - 1;  // clamp: discarded rows read valid memory

  auto stage = [&](int s, int p) {
    const int k0 = s << 5;
    {  // weights -> At[p][row][K]
      v8bf t0, t1;
      if (k0 + 32 <= Cin) {  // uniform scalar branch (false only when Cin==3)
        const float* src = &W[(size_t)rg * Cin + k0 + wkb];
#pragma unroll
        for (int j = 0; j < 8; ++j) { t0[j] = tobf(src[j]); t1[j] = tobf(src[8 + j]); }
      } else {  // clamped + select tail (W is a real input buffer)
#pragma unroll
        for (int j = 0; j < 8; ++j) {
          int Ka = k0 + wkb + j, Kb2 = Ka + 8;
          int Kca = Ka < Cin ? Ka : Cin - 1;
          int Kcb = Kb2 < Cin ? Kb2 : Cin - 1;
          float la = W[(size_t)rg * Cin + Kca];
          float lb = W[(size_t)rg * Cin + Kcb];
          t0[j] = tobf(Ka < Cin ? la : 0.f);
          t1[j] = tobf(Kb2 < Cin ? lb : 0.f);
        }
      }
      *(v8bf*)&At[p][wr][wkb] = t0;
      *(v8bf*)&At[p][wr][wkb + 8] = t1;
    }
    {  // activations (transposed) -> Bt[p][col][K]; X is a workspace slab
      int c = k0 + sk;
      bool cv = (c < Cin);
      int col = m0 + scol;
      float xv[8];
      if (stride == 1) {
        const float* src = &X[(size_t)(n * Cin + c) * M + col];
#pragma unroll
        for (int j = 0; j < 8; ++j) { float ld = src[j]; xv[j] = cv ? ld : 0.f; }
      } else {
        const float* src = &X[(size_t)(n * Cin + c) * Tin * VJ];
#pragma unroll
        for (int j = 0; j < 8; ++j) {
          int cl = col + j;
          int ti = cl / VJ, vi = cl - ti * VJ;
          float ld = src[(size_t)ti * stride * VJ + vi];
          xv[j] = cv ? ld : 0.f;
        }
      }
#pragma unroll
      for (int j = 0; j < 8; ++j) Bt[p][scol + j][sk] = tobf(xv[j]);
    }
  };

  v8f acc[4] = {};
  stage(0, 0);
  __syncthreads();
  for (int s = 0; s < steps; ++s) {
    const int p = s & 1;
    v8bf alo = *(const v8bf*)&At[p][arow][kh];
    v8bf ahi = *(const v8bf*)&At[p][arow][kh + 16];
    v16bf a = __builtin_shufflevector(alo, ahi, 0, 1, 2, 3, 4, 5, 6, 7, 8, 9, 10, 11, 12, 13, 14, 15);
#pragma unroll
    for (int tt = 0; tt < 4; ++tt) {
      v8bf blo = *(const v8bf*)&Bt[p][tt * 16 + bcol][bkb];
      v8bf bhi = *(const v8bf*)&Bt[p][tt * 16 + bcol][bkb + 16];
      v16bf b = __builtin_shufflevector(blo, bhi, 0, 1, 2, 3, 4, 5, 6, 7, 8, 9, 10, 11, 12, 13, 14, 15);
      acc[tt] = __builtin_amdgcn_wmma_f32_16x16x32_bf16(false, a, false, b, (short)0,
                                                        acc[tt], false, false);
    }
    if (s + 1 < steps) stage(s + 1, 1 - p);
    __syncthreads();
  }

#pragma unroll
  for (int tt = 0; tt < 4; ++tt) {
    int col = m0 + tt * 16 + (lane & 15);
    int t = col / VJ, v = col - t * VJ;
#pragma unroll
    for (int r = 0; r < 8; ++r) {
      int o = o0 + (wave << 4) + r + ((lane & 16) ? 8 : 0);
      if (o < O) {
        float val = acc[tt][r] + bias[o];
        if (bg) val = (val - bm[o]) * (bg[o] * rsqrtf(bv[o] + EPSB)) + bb[o];
        Y[((size_t)(n * O + o) * Tout + t) * VJ + v] = val;
      }
    }
  }
}

// ---------------------------------------------------------------- CTR combine + BN1 + ReLU
__global__ void ctr_combine_k(const float* __restrict__ X3, const float* __restrict__ X4,
                              const float* __restrict__ AL, const float* __restrict__ Am,
                              const float* __restrict__ g1, const float* __restrict__ b1,
                              const float* __restrict__ m1, const float* __restrict__ v1,
                              float* __restrict__ H, int N, int Cout, int inter, int T) {
  size_t idx = (size_t)blockIdx.x * blockDim.x + threadIdx.x;
  size_t total = (size_t)N * Cout * T;
  if (idx >= total) return;
  int t = (int)(idx % T);
  int o = (int)((idx / T) % Cout);
  int n = (int)(idx / ((size_t)T * Cout));
  size_t base = ((size_t)(n * Cout + o) * T + t) * VJ;
  float x3r[VJ], x4r[VJ];
#pragma unroll
  for (int v = 0; v < VJ; ++v) { x3r[v] = X3[base + v]; x4r[v] = X4[base + v]; }
  const float* al = AL + ((size_t)n * inter + (o % inter)) * VJ;
  float s = 0.f;
#pragma unroll
  for (int v = 0; v < VJ; ++v) s += al[v] * x3r[v];
  float scale = g1[o] * rsqrtf(v1[o] + EPSB), mu = m1[o], beta = b1[o];
#pragma unroll
  for (int u = 0; u < VJ; ++u) {
    float gsum = 0.f;
#pragma unroll
    for (int v = 0; v < VJ; ++v) gsum += Am[u * VJ + v] * x4r[v];
    float y = (s + gsum - mu) * scale + beta;
    H[base + u] = y > 0.f ? y : 0.f;
  }
}

// ---------------------------------------------------------------- temporal conv (WMMA bf16)
// Out = relu(BN2( sum_{ci,kt} TW[o,ci,kt]*H[n,ci,to*s+kt-4,v] + tb[o]) + RES)
// 9 shifted 1x1 GEMMs chained into 4 v8f tiles (4*9*C/32 WMMAs per tile set).
__global__ __launch_bounds__(256)
void tconv_wmma(const float* __restrict__ H, const float* __restrict__ TW,
                const float* __restrict__ tbias,
                const float* __restrict__ g2, const float* __restrict__ b2,
                const float* __restrict__ m2, const float* __restrict__ v2,
                const float* __restrict__ RES, float* __restrict__ Out,
                int N, int C, int Tin, int Tout, int stride, int Mtiles) {
  __shared__ __attribute__((aligned(64))) __bf16 At[2][128][40];
  __shared__ __attribute__((aligned(64))) __bf16 Bt[2][64][40];
  const int tid = threadIdx.x, lane = tid & 31, wave = tid >> 5;
  const int n = blockIdx.x / Mtiles;
  const int m0 = (blockIdx.x % Mtiles) * 64;
  const int o0 = blockIdx.y * 128;
  const int M = Tout * VJ;
  const int Min = Tin * VJ;
  const int arow = (wave << 4) + (lane & 15);
  const int kh = (lane & 16) ? 8 : 0;
  const int bkb = (lane & 16) ? 8 : 0;
  const int bcol = lane & 15;
  const int wr = tid >> 1;
  const int wkb = (tid & 1) * 16;
  const int sk = tid >> 3;
  const int scol = (tid & 7) * 8;
  const int c32 = C >> 5;
  const int steps = 9 * c32;

  int rg = o0 + wr;
  if (rg > C - 1) rg = C - 1;  // clamp (discarded rows)

  auto stage = [&](int s, int p) {
    const int kt = s / c32;
    const int c0 = (s - kt * c32) << 5;
    {  // weights: K = ci (stride-9 gather), contiguous LDS stores
      v8bf t0, t1;
      const float* src = &TW[((size_t)rg * C + c0 + wkb) * 9 + kt];
#pragma unroll
      for (int j = 0; j < 8; ++j) {
        t0[j] = tobf(src[(size_t)j * 9]);
        t1[j] = tobf(src[(size_t)(j + 8) * 9]);
      }
      *(v8bf*)&At[p][wr][wkb] = t0;
      *(v8bf*)&At[p][wr][wkb + 8] = t1;
    }
    {  // activations: temporal zero-padding via selects; H is an interior slab
      const int c = c0 + sk;  // always < C (C multiple of 32)
      const float* src = &H[(size_t)(n * C + c) * Min];
      const int col = m0 + scol;
      float xv[8];
      if (stride == 1) {
        int q = col + (kt - 4) * VJ;  // may be slightly OOB; stays inside d_ws
        const float* s2 = src + q;
#pragma unroll
        for (int j = 0; j < 8; ++j) {
          float ld = s2[j];
          xv[j] = ((unsigned)(q + j) < (unsigned)Min) ? ld : 0.f;
        }
      } else {
#pragma unroll
        for (int j = 0; j < 8; ++j) {
          int cl = col + j;
          int to = cl / VJ, vi = cl - to * VJ;
          int ti = to * stride + kt - 4;
          float ld = src[(ptrdiff_t)ti * VJ + vi];
          xv[j] = ((unsigned)ti < (unsigned)Tin) ? ld : 0.f;
        }
      }
#pragma unroll
      for (int j = 0; j < 8; ++j) Bt[p][scol + j][sk] = tobf(xv[j]);
    }
  };

  v8f acc[4] = {};
  stage(0, 0);
  __syncthreads();
  for (int s = 0; s < steps; ++s) {
    const int p = s & 1;
    v8bf alo = *(const v8bf*)&At[p][arow][kh];
    v8bf ahi = *(const v8bf*)&At[p][arow][kh + 16];
    v16bf a = __builtin_shufflevector(alo, ahi, 0, 1, 2, 3, 4, 5, 6, 7, 8, 9, 10, 11, 12, 13, 14, 15);
#pragma unroll
    for (int tt = 0; tt < 4; ++tt) {
      v8bf blo = *(const v8bf*)&Bt[p][tt * 16 + bcol][bkb];
      v8bf bhi = *(const v8bf*)&Bt[p][tt * 16 + bcol][bkb + 16];
      v16bf b = __builtin_shufflevector(blo, bhi, 0, 1, 2, 3, 4, 5, 6, 7, 8, 9, 10, 11, 12, 13, 14, 15);
      acc[tt] = __builtin_amdgcn_wmma_f32_16x16x32_bf16(false, a, false, b, (short)0,
                                                        acc[tt], false, false);
    }
    if (s + 1 < steps) stage(s + 1, 1 - p);
    __syncthreads();
  }

#pragma unroll
  for (int tt = 0; tt < 4; ++tt) {
    int col = m0 + tt * 16 + (lane & 15);
    int to = col / VJ, v = col - to * VJ;
#pragma unroll
    for (int r = 0; r < 8; ++r) {
      int o = o0 + (wave << 4) + r + ((lane & 16) ? 8 : 0);
      if (o < C) {
        size_t oidx = ((size_t)(n * C + o) * Tout + to) * VJ + v;
        float val = acc[tt][r] + tbias[o];
        val = (val - m2[o]) * (g2[o] * rsqrtf(v2[o] + EPSB)) + b2[o];
        val += RES[oidx];
        Out[oidx] = val > 0.f ? val : 0.f;
      }
    }
  }
}

// ---------------------------------------------------------------- GAP + FC
__global__ void gap_k(const float* __restrict__ X, float* __restrict__ P, int N, int C,
                      int T) {
  int idx = blockIdx.x * blockDim.x + threadIdx.x;
  if (idx >= N * C) return;
  float s = 0.f;
  size_t base = (size_t)idx * T * VJ;
  for (int i = 0; i < T * VJ; ++i) s += X[base + i];
  P[idx] = s / (float)(T * VJ);
}

__global__ void fc_k(const float* __restrict__ P, const float* __restrict__ FW,
                     const float* __restrict__ FB, float* __restrict__ O, int N, int C,
                     int K) {
  int idx = blockIdx.x * blockDim.x + threadIdx.x;
  if (idx >= N * K) return;
  int k = idx % K, n = idx / K;
  float s = FB[k];
  for (int o = 0; o < C; ++o) s += P[n * C + o] * FW[o * K + k];
  O[idx] = s;
}

// ---------------------------------------------------------------- host orchestration
extern "C" void kernel_launch(void* const* d_in, const int* in_sizes, int n_in,
                              void* d_out, int out_size, void* d_ws, size_t ws_size,
                              hipStream_t stream) {
  (void)in_sizes; (void)n_in; (void)out_size; (void)ws_size;
  const int N = 128, CIN = 3, T0 = 256, NC = 23;
  static const int CFG[6][3] = {{3, 64, 1},  {64, 64, 1},  {64, 128, 2},
                                {128, 128, 1}, {128, 256, 2}, {256, 256, 1}};
  static const int TIN[6] = {256, 256, 256, 128, 128, 64};

  int idx = 0;
  const float* x = (const float*)d_in[idx++];
  const float* dbn[4];
  for (int i = 0; i < 4; ++i) dbn[i] = (const float*)d_in[idx++];

  struct Blk {
    const float *w1, *b1, *w2, *b2, *w3, *b3, *w4, *b4;
    const float* bn1[4];
    const float *tw, *tb;
    const float* bn2[4];
    const float *rw, *rb;
    const float* rbn[4];
    bool id;
  } B[6];

  for (int bi = 0; bi < 6; ++bi) {
    Blk& p = B[bi];
    p.id = (CFG[bi][0] == CFG[bi][1] && CFG[bi][2] == 1);
    p.w1 = (const float*)d_in[idx++]; p.b1 = (const float*)d_in[idx++];
    p.w2 = (const float*)d_in[idx++]; p.b2 = (const float*)d_in[idx++];
    p.w3 = (const float*)d_in[idx++]; p.b3 = (const float*)d_in[idx++];
    p.w4 = (const float*)d_in[idx++]; p.b4 = (const float*)d_in[idx++];
    for (int i = 0; i < 4; ++i) p.bn1[i] = (const float*)d_in[idx++];
    p.tw = (const float*)d_in[idx++]; p.tb = (const float*)d_in[idx++];
    for (int i = 0; i < 4; ++i) p.bn2[i] = (const float*)d_in[idx++];
    if (!p.id) {
      p.rw = (const float*)d_in[idx++]; p.rb = (const float*)d_in[idx++];
      for (int i = 0; i < 4; ++i) p.rbn[i] = (const float*)d_in[idx++];
    } else {
      p.rw = p.rb = nullptr;
      for (int i = 0; i < 4; ++i) p.rbn[i] = nullptr;
    }
  }
  const float* fcw = (const float*)d_in[idx++];
  const float* fcb = (const float*)d_in[idx++];

  const size_t SLAB = (size_t)128 * 64 * 256 * 17;
  float* W = (float*)d_ws;
  float* S0 = W;
  float* S1 = W + SLAB;
  float* S2 = W + 2 * SLAB;
  float* S3 = W + 3 * SLAB;
  float* S4 = W + 4 * SLAB;
  float* S5 = W + 5 * SLAB;
  float* xm = W + 6 * SLAB;
  float* al = xm + (size_t)128 * 256 * VJ;
  float* Am = al + (size_t)128 * 64 * VJ;
  float* pool = Am + VJ * VJ;

  adj_k<<<1, 32, 0, stream>>>(Am);
  {
    size_t tot = (size_t)N * CIN * T0 * VJ;
    data_bn_k<<<(unsigned)((tot + 255) / 256), 256, 0, stream>>>(
        x, dbn[0], dbn[1], dbn[2], dbn[3], S0, N, CIN, T0);
  }

  float* xcur = S0;
  float* xnext = S5;
  for (int bi = 0; bi < 6; ++bi) {
    const int cin = CFG[bi][0], cout = CFG[bi][1], s = CFG[bi][2];
    const int Tin = TIN[bi], Tout = Tin / s, inter = cout / 4;
    Blk& p = B[bi];

    mean_t_k<<<(N * cin * VJ + 255) / 256, 256, 0, stream>>>(xcur, xm, N, cin, Tin);
    alpha_k<<<(N * inter + 255) / 256, 256, 0, stream>>>(xm, p.w2, p.b2, al, N, cin,
                                                         inter);

    const int Mi = Tin * VJ, Mti = Mi / 64;  // exact
    dim3 g1((unsigned)(N * Mti), (unsigned)((cout + 127) / 128));
    conv1x1_wmma<<<g1, 256, 0, stream>>>(xcur, p.w3, p.b3, nullptr, nullptr, nullptr,
                                         nullptr, S1, N, cin, Tin, Tin, cout, 1, Mti);
    conv1x1_wmma<<<g1, 256, 0, stream>>>(xcur, p.w4, p.b4, nullptr, nullptr, nullptr,
                                         nullptr, S2, N, cin, Tin, Tin, cout, 1, Mti);

    {
      size_t tot = (size_t)N * cout * Tin;
      ctr_combine_k<<<(unsigned)((tot + 255) / 256), 256, 0, stream>>>(
          S1, S2, al, Am, p.bn1[0], p.bn1[1], p.bn1[2], p.bn1[3], S3, N, cout, inter,
          Tin);
    }

    const float* res = xcur;
    const int Mo = Tout * VJ, Mto = Mo / 64;  // exact
    dim3 go((unsigned)(N * Mto), (unsigned)((cout + 127) / 128));
    if (!p.id) {
      conv1x1_wmma<<<go, 256, 0, stream>>>(xcur, p.rw, p.rb, p.rbn[0], p.rbn[1],
                                           p.rbn[2], p.rbn[3], S4, N, cin, Tin, Tout,
                                           cout, s, Mto);
      res = S4;
    }

    tconv_wmma<<<go, 256, 0, stream>>>(S3, p.tw, p.tb, p.bn2[0], p.bn2[1], p.bn2[2],
                                       p.bn2[3], res, xnext, N, cout, Tin, Tout, s,
                                       Mto);

    float* tmp = xcur; xcur = xnext; xnext = tmp;
  }

  gap_k<<<(N * 256 + 255) / 256, 256, 0, stream>>>(xcur, pool, N, 256, 64);
  fc_k<<<(N * NC + 255) / 256, 256, 0, stream>>>(pool, fcw, fcb, (float*)d_out, N, 256,
                                                 NC);
}
